// EncoderBlock_61632780697696
// MI455X (gfx1250) — matrix operand order repaired and checked by reading, not compiled, not run on hardware
//
#include <hip/hip_runtime.h>

// ---------------- problem constants ----------------
#define Bq   2
#define Sq   2048
#define Dq   1024
#define Hq   16
#define DKq  64
#define DFFq 4096
#define WINq 128
#define GLOq 16

typedef __attribute__((ext_vector_type(16))) __bf16 v16bf;
typedef __attribute__((ext_vector_type(8)))  float  v8f;

union FragBF { v16bf v; uint4 q[2]; };

#if defined(__gfx1250__) && __has_builtin(__builtin_amdgcn_global_load_async_to_lds_b128)
#define USE_ASYNC_LDS 1
typedef int v4i_ __attribute__((vector_size(4 * sizeof(int))));
typedef __attribute__((address_space(1))) v4i_* gasp;   // global addrspace ptr
typedef __attribute__((address_space(3))) v4i_* lasp;   // LDS addrspace ptr

__device__ __forceinline__ void async_cp16(const unsigned short* g, unsigned short* l) {
    unsigned short* gg = const_cast<unsigned short*>(g);
    __builtin_amdgcn_global_load_async_to_lds_b128((gasp)gg, (lasp)l, 0, 0);
}
#else
#define USE_ASYNC_LDS 0
#endif

__device__ __forceinline__ unsigned short f2bf(float f) {
    unsigned int u = __float_as_uint(f);
    unsigned int r = u + 0x7FFFu + ((u >> 16) & 1u);
    return (unsigned short)(r >> 16);
}

__device__ __forceinline__ v8f zero8() {
    v8f z;
    #pragma unroll
    for (int i = 0; i < 8; ++i) z[i] = 0.f;
    return z;
}

__device__ __forceinline__ v8f wmma_bf16(const FragBF& a, const FragBF& b, v8f c) {
    return __builtin_amdgcn_wmma_f32_16x16x32_bf16(false, a.v, false, b.v,
                                                   (short)0, c, false, false);
}

__device__ __forceinline__ float rmax16(float v) {
    #pragma unroll
    for (int off = 1; off < 16; off <<= 1) v = fmaxf(v, __shfl_xor(v, off, 32));
    return v;
}
__device__ __forceinline__ float rsum16(float v) {
    #pragma unroll
    for (int off = 1; off < 16; off <<= 1) v += __shfl_xor(v, off, 32);
    return v;
}

// ---------------- LayerNorm (f32 in, bf16 out) ----------------
__global__ __launch_bounds__(256) void ln_kernel(const float* __restrict__ x,
                                                 const float* __restrict__ g,
                                                 const float* __restrict__ b,
                                                 unsigned short* __restrict__ out) {
    const int row = blockIdx.x;
    const int tid = threadIdx.x;
    const float* xr = x + (size_t)row * Dq;

    float v[4], s = 0.f, s2 = 0.f;
    #pragma unroll
    for (int k = 0; k < 4; ++k) {
        v[k] = xr[tid + k * 256];
        s += v[k]; s2 += v[k] * v[k];
    }
    #pragma unroll
    for (int off = 1; off < 32; off <<= 1) {
        s  += __shfl_xor(s,  off, 32);
        s2 += __shfl_xor(s2, off, 32);
    }
    __shared__ float rs[8], rs2[8];
    const int wv = tid >> 5, lane = tid & 31;
    if (lane == 0) { rs[wv] = s; rs2[wv] = s2; }
    __syncthreads();
    float S1 = 0.f, S2 = 0.f;
    #pragma unroll
    for (int i = 0; i < 8; ++i) { S1 += rs[i]; S2 += rs2[i]; }
    const float mu  = S1 * (1.f / Dq);
    const float var = S2 * (1.f / Dq) - mu * mu;
    const float inv = rsqrtf(var + 1e-5f);
    #pragma unroll
    for (int k = 0; k < 4; ++k) {
        const int c = tid + k * 256;
        out[(size_t)row * Dq + c] = f2bf((v[k] - mu) * inv * g[c] + b[c]);
    }
}

// ---------------- weight f32 [K][N] -> bf16 transposed [N][K] ----------------
__global__ __launch_bounds__(256) void wtr_kernel(const float* __restrict__ in,
                                                  unsigned short* __restrict__ out,
                                                  int K, int N) {
    const size_t total = (size_t)K * N;
    for (size_t idx = blockIdx.x * 256ull + threadIdx.x; idx < total;
         idx += (size_t)gridDim.x * 256ull) {
        const size_t n = idx / (size_t)K;
        const size_t k = idx - n * (size_t)K;
        out[idx] = f2bf(in[k * (size_t)N + n]);
    }
}

// ---------------- V bf16 [B*S][D] -> [B,H,DK,S] ----------------
__global__ __launch_bounds__(256) void vtr_kernel(const unsigned short* __restrict__ v,
                                                  unsigned short* __restrict__ vt) {
    const size_t total = (size_t)Bq * Sq * Dq;
    for (size_t o = blockIdx.x * 256ull + threadIdx.x; o < total;
         o += (size_t)gridDim.x * 256ull) {
        const int s  = (int)(o % Sq);
        const size_t t = o / Sq;
        const int d  = (int)(t % DKq);
        const int bh = (int)(t / DKq);
        const int h  = bh % Hq;
        const int bb = bh / Hq;
        vt[o] = v[((size_t)bb * Sq + s) * Dq + h * DKq + d];
    }
}

// ---- bf16 GEMM: C[M,N] = A[M,K] @ Bt[N,K]^T + bias (+gelu)(+res), 128x128 tile ----
__global__ __launch_bounds__(256) void gemm_bf16(const unsigned short* __restrict__ A,
                                                 const unsigned short* __restrict__ Bt,
                                                 const float* __restrict__ bias,
                                                 const float* __restrict__ res,
                                                 float* __restrict__ outF,
                                                 unsigned short* __restrict__ outB,
                                                 int M, int N, int K, int act) {
    // double-buffered LDS: 2 x (128x32) bf16 for A and for B = 32 KB total
    __shared__ alignas(16) unsigned short As[2 * 128 * 32];
    __shared__ alignas(16) unsigned short Bs[2 * 128 * 32];

    const int tid  = threadIdx.x;
    const int lane = tid & 31, wv = tid >> 5;
    const int wm = wv & 3, wn = wv >> 2;      // 4 M-waves x 2 N-waves
    const int g = lane >> 4, hl = lane & 15;
    const int bm = blockIdx.y, bn = blockIdx.x;

    v8f acc[2][4];
    #pragma unroll
    for (int mt = 0; mt < 2; ++mt)
        #pragma unroll
        for (int nt = 0; nt < 4; ++nt) acc[mt][nt] = zero8();

    // staging: 128 rows x 32 cols, 256 threads -> 16 elements (32 B) each
    const int sRow = tid >> 1, sCol = (tid & 1) * 16;
    const unsigned short* gA = A  + (size_t)(bm * 128 + sRow) * K + sCol;
    const unsigned short* gB = Bt + (size_t)(bn * 128 + sRow) * K + sCol;
    unsigned short* const aDst = As + sRow * 32 + sCol;
    unsigned short* const bDst = Bs + sRow * 32 + sCol;

    auto stage = [&](int buf, int k0) {
#if USE_ASYNC_LDS
        async_cp16(gA + k0,     aDst + buf * 4096);
        async_cp16(gA + k0 + 8, aDst + buf * 4096 + 8);
        async_cp16(gB + k0,     bDst + buf * 4096);
        async_cp16(gB + k0 + 8, bDst + buf * 4096 + 8);
#else
        const uint4* pa = (const uint4*)(gA + k0);
        uint4 a0 = pa[0], a1 = pa[1];
        *(uint4*)(aDst + buf * 4096)     = a0;
        *(uint4*)(aDst + buf * 4096 + 8) = a1;
        const uint4* pb = (const uint4*)(gB + k0);
        uint4 b0 = pb[0], b1 = pb[1];
        *(uint4*)(bDst + buf * 4096)     = b0;
        *(uint4*)(bDst + buf * 4096 + 8) = b1;
#endif
    };

    const int nIter = K >> 5;
    stage(0, 0);

    for (int it = 0; it < nIter; ++it) {
        const int cur = it & 1;
#if USE_ASYNC_LDS
        asm volatile("s_wait_asynccnt 0x0" ::: "memory");
#endif
        __syncthreads();
        if (it + 1 < nIter) {
            stage(1 - cur, (it + 1) << 5);
            __builtin_prefetch(gA + ((it + 2) << 5), 0, 3);
            __builtin_prefetch(gB + ((it + 2) << 5), 0, 3);
        }

        const unsigned short* as = As + cur * 4096;
        const unsigned short* bs = Bs + cur * 4096;
        FragBF af[2], bf[4];
        #pragma unroll
        for (int mt = 0; mt < 2; ++mt) {
            const int r = wm * 32 + mt * 16 + hl;
            af[mt].q[0] = *(const uint4*)(as + r * 32 + g * 8);
            af[mt].q[1] = *(const uint4*)(as + r * 32 + 16 + g * 8);
        }
        #pragma unroll
        for (int nt = 0; nt < 4; ++nt) {
            const int c = wn * 64 + nt * 16 + hl;
            bf[nt].q[0] = *(const uint4*)(bs + c * 32 + g * 8);
            bf[nt].q[1] = *(const uint4*)(bs + c * 32 + 16 + g * 8);
        }
        #pragma unroll
        for (int mt = 0; mt < 2; ++mt)
            #pragma unroll
            for (int nt = 0; nt < 4; ++nt)
                acc[mt][nt] = wmma_bf16(af[mt], bf[nt], acc[mt][nt]);
    }

    // epilogue
    #pragma unroll
    for (int mt = 0; mt < 2; ++mt) {
        #pragma unroll
        for (int nt = 0; nt < 4; ++nt) {
            #pragma unroll
            for (int r = 0; r < 8; ++r) {
                const int grow = bm * 128 + wm * 32 + mt * 16 + r + (g ? 8 : 0);
                const int gcol = bn * 128 + wn * 64 + nt * 16 + hl;
                float v = acc[mt][nt][r] + bias[gcol];
                if (act) v = 0.5f * v * (1.0f + erff(v * 0.70710678118654752f));
                if (res) v += res[(size_t)grow * N + gcol];
                if (outF) outF[(size_t)grow * N + gcol] = v;
                else      outB[(size_t)grow * N + gcol] = f2bf(v);
            }
        }
    }
}

// ---------------- local+global attention, flash-style, one wave per 16-query tile ---
__global__ __launch_bounds__(256) void attn_kernel(const unsigned short* __restrict__ qb,
                                                   const unsigned short* __restrict__ kb,
                                                   const unsigned short* __restrict__ vt,
                                                   unsigned short* __restrict__ ob) {
    __shared__ alignas(16) unsigned short Pl[8 * 16 * 32]; // per-wave 16x32 bf16 P patch

    const int tid  = threadIdx.x;
    const int lane = tid & 31, wv = tid >> 5;
    const int g = lane >> 4, hl = lane & 15;

    const int wid = blockIdx.x * 8 + wv;            // global q-tile id
    const int b   = wid >> 11;                      // / (H * S/16)
    const int rem = wid & 2047;
    const int h   = rem >> 7;
    const int qt  = rem & 127;
    const int q0  = qt * 16;

    unsigned short* P = Pl + wv * 512;

    // Q fragments for the two k=32 steps of DK=64 (A-matrix layout)
    FragBF qa[2];
    const unsigned short* qrow = qb + ((size_t)(b * Sq + q0 + hl)) * Dq + h * DKq;
    qa[0].q[0] = *(const uint4*)(qrow + g * 16);
    qa[0].q[1] = *(const uint4*)(qrow + g * 16 + 8);
    qa[1].q[0] = *(const uint4*)(qrow + 32 + g * 16);
    qa[1].q[1] = *(const uint4*)(qrow + 40 + g * 16);

    v8f o[4];
    #pragma unroll
    for (int t = 0; t < 4; ++t) o[t] = zero8();
    float m[8], l[8];
    #pragma unroll
    for (int r = 0; r < 8; ++r) { m[r] = -3.0e38f; l[r] = 0.f; }

    // key-tile set: {0} U [qt-8, qt+8]; q-tile 0 holds global queries -> all tiles
    int lo = (qt > 8) ? qt - 8 : 0;
    int hi = (qt + 8 > 127) ? 127 : qt + 8;
    if (qt == 0) { lo = 0; hi = 127; }
    const int extra = (lo > 0) ? 1 : 0;
    const int count = hi - lo + 1 + extra;

    for (int i = 0; i < count; i += 2) {
        const int ta  = (extra && i == 0) ? 0 : (lo + i - extra);
        const int tb  = (i + 1 < count) ? (lo + i + 1 - extra) : -1;
        const int tbl = (tb < 0) ? ta : tb;

        // scores for the two 16-key tiles
        v8f c0 = zero8(), c1 = zero8();
        #pragma unroll
        for (int ks = 0; ks < 2; ++ks) {
            FragBF kf;
            const unsigned short* kr =
                kb + ((size_t)(b * Sq + ta * 16 + hl)) * Dq + h * DKq + ks * 32 + g * 16;
            kf.q[0] = *(const uint4*)(kr);
            kf.q[1] = *(const uint4*)(kr + 8);
            c0 = wmma_bf16(qa[ks], kf, c0);
        }
        #pragma unroll
        for (int ks = 0; ks < 2; ++ks) {
            FragBF kf;
            const unsigned short* kr =
                kb + ((size_t)(b * Sq + tbl * 16 + hl)) * Dq + h * DKq + ks * 32 + g * 16;
            kf.q[0] = *(const uint4*)(kr);
            kf.q[1] = *(const uint4*)(kr + 8);
            c1 = wmma_bf16(qa[ks], kf, c1);
        }

        // online softmax update (per row r; lanes 0-15 -> M=r, lanes 16-31 -> M=r+8)
        float pa[8], pb[8];
        #pragma unroll
        for (int r = 0; r < 8; ++r) {
            const int M  = r + (g ? 8 : 0);
            const int ii = q0 + M;
            const int ja = ta * 16 + hl;
            const int jb = tb * 16 + hl;
            const int da = (ii > ja) ? ii - ja : ja - ii;
            const int db = (ii > jb) ? ii - jb : jb - ii;
            const bool okA = (da <= WINq) | (ii < GLOq) | (ja < GLOq);
            const bool okB = (tb >= 0) & ((db <= WINq) | (ii < GLOq) | (jb < GLOq));
            const float sa = okA ? c0[r] * 0.125f : -3.0e38f;
            const float sb = okB ? c1[r] * 0.125f : -3.0e38f;

            float mx = rmax16(fmaxf(sa, sb));
            const float mnew  = fmaxf(m[r], mx);
            const float alpha = __expf(m[r] - mnew);
            pa[r] = okA ? __expf(sa - mnew) : 0.f;
            pb[r] = okB ? __expf(sb - mnew) : 0.f;
            float rs = rsum16(pa[r] + pb[r]);
            l[r] = l[r] * alpha + rs;
            m[r] = mnew;
            #pragma unroll
            for (int t = 0; t < 4; ++t) o[t][r] *= alpha;
        }

        // transpose P (C-layout -> A-layout) via per-wave LDS patch
        #pragma unroll
        for (int r = 0; r < 8; ++r) {
            const int M = r + (g ? 8 : 0);
            P[M * 32 + hl]      = f2bf(pa[r]);
            P[M * 32 + 16 + hl] = f2bf(pb[r]);
        }
        asm volatile("s_wait_dscnt 0x0" ::: "memory");

        FragBF pf;
        pf.q[0] = *(const uint4*)(P + hl * 32 + g * 16);
        pf.q[1] = *(const uint4*)(P + hl * 32 + g * 16 + 8);

        // P @ V : V^T laid out [B,H,DK,S]; lane-half g selects key tile a/b
        const int keyBase = (g ? tbl : ta) * 16;
        #pragma unroll
        for (int t = 0; t < 4; ++t) {
            FragBF vf;
            const unsigned short* vr =
                vt + ((size_t)((b * Hq + h) * DKq + t * 16 + hl)) * Sq + keyBase;
            vf.q[0] = *(const uint4*)(vr);
            vf.q[1] = *(const uint4*)(vr + 8);
            o[t] = wmma_bf16(pf, vf, o[t]);
        }
    }

    // write normalized output as bf16
    #pragma unroll
    for (int t = 0; t < 4; ++t) {
        #pragma unroll
        for (int r = 0; r < 8; ++r) {
            const int M = r + (g ? 8 : 0);
            const float v = o[t][r] / fmaxf(l[r], 1e-20f);
            ob[((size_t)(b * Sq + q0 + M)) * Dq + h * DKq + t * 16 + hl] = f2bf(v);
        }
    }
}

// ---------------- host-side launcher ----------------
extern "C" void kernel_launch(void* const* d_in, const int* in_sizes, int n_in,
                              void* d_out, int out_size, void* d_ws, size_t ws_size,
                              hipStream_t stream) {
    const float* x    = (const float*)d_in[0];
    const float* Wq   = (const float*)d_in[1];
    const float* bq   = (const float*)d_in[2];
    const float* Wk   = (const float*)d_in[3];
    const float* bk   = (const float*)d_in[4];
    const float* Wv   = (const float*)d_in[5];
    const float* bv   = (const float*)d_in[6];
    const float* Wo   = (const float*)d_in[7];
    const float* bo   = (const float*)d_in[8];
    const float* ln1g = (const float*)d_in[9];
    const float* ln1b = (const float*)d_in[10];
    const float* W1   = (const float*)d_in[11];
    const float* b1   = (const float*)d_in[12];
    const float* W2   = (const float*)d_in[13];
    const float* b2   = (const float*)d_in[14];
    const float* ln2g = (const float*)d_in[15];
    const float* ln2b = (const float*)d_in[16];

    const int M = Bq * Sq;  // 4096 rows

    unsigned char* w = (unsigned char*)d_ws;
    auto carve = [&](size_t bytes) -> void* {
        void* p = (void*)w;
        w += (bytes + 255) & ~(size_t)255;
        return p;
    };
    unsigned short* Wqt  = (unsigned short*)carve((size_t)Dq * Dq * 2);
    unsigned short* Wkt  = (unsigned short*)carve((size_t)Dq * Dq * 2);
    unsigned short* Wvt  = (unsigned short*)carve((size_t)Dq * Dq * 2);
    unsigned short* Wot  = (unsigned short*)carve((size_t)Dq * Dq * 2);
    unsigned short* W1t  = (unsigned short*)carve((size_t)Dq * DFFq * 2);
    unsigned short* W2t  = (unsigned short*)carve((size_t)Dq * DFFq * 2);
    unsigned short* xnB  = (unsigned short*)carve((size_t)M * Dq * 2);
    unsigned short* qB   = (unsigned short*)carve((size_t)M * Dq * 2);
    unsigned short* kB   = (unsigned short*)carve((size_t)M * Dq * 2);
    unsigned short* vB   = (unsigned short*)carve((size_t)M * Dq * 2);
    unsigned short* vtB  = (unsigned short*)carve((size_t)M * Dq * 2);
    unsigned short* aoB  = (unsigned short*)carve((size_t)M * Dq * 2);
    float*          x1F  = (float*)         carve((size_t)M * Dq * 4);
    unsigned short* xn2B = (unsigned short*)carve((size_t)M * Dq * 2);
    unsigned short* hB   = (unsigned short*)carve((size_t)M * DFFq * 2);

    // 1) weight converts (f32 [K][N] -> bf16 [N][K])
    wtr_kernel<<<4096, 256, 0, stream>>>(Wq, Wqt, Dq, Dq);
    wtr_kernel<<<4096, 256, 0, stream>>>(Wk, Wkt, Dq, Dq);
    wtr_kernel<<<4096, 256, 0, stream>>>(Wv, Wvt, Dq, Dq);
    wtr_kernel<<<4096, 256, 0, stream>>>(Wo, Wot, Dq, Dq);
    wtr_kernel<<<16384, 256, 0, stream>>>(W1, W1t, Dq, DFFq);  // -> [DFF][D]
    wtr_kernel<<<16384, 256, 0, stream>>>(W2, W2t, DFFq, Dq);  // -> [D][DFF]

    // 2) LN1
    ln_kernel<<<M, 256, 0, stream>>>(x, ln1g, ln1b, xnB);

    // 3) QKV projections (bf16 out)
    dim3 gQKV(Dq / 128, M / 128);
    gemm_bf16<<<gQKV, 256, 0, stream>>>(xnB, Wqt, bq, nullptr, nullptr, qB, M, Dq, Dq, 0);
    gemm_bf16<<<gQKV, 256, 0, stream>>>(xnB, Wkt, bk, nullptr, nullptr, kB, M, Dq, Dq, 0);
    gemm_bf16<<<gQKV, 256, 0, stream>>>(xnB, Wvt, bv, nullptr, nullptr, vB, M, Dq, Dq, 0);

    // 4) V transpose for P@V B-side
    vtr_kernel<<<32768, 256, 0, stream>>>(vB, vtB);

    // 5) attention
    attn_kernel<<<(Bq * Hq * (Sq / 16)) / 8, 256, 0, stream>>>(qB, kB, vtB, aoB);

    // 6) out-proj + residual
    gemm_bf16<<<gQKV, 256, 0, stream>>>(aoB, Wot, bo, x, x1F, nullptr, M, Dq, Dq, 0);

    // 7) LN2
    ln_kernel<<<M, 256, 0, stream>>>(x1F, ln2g, ln2b, xn2B);

    // 8) FFN1 + exact GELU (bf16 out)
    dim3 gF1(DFFq / 128, M / 128);
    gemm_bf16<<<gF1, 256, 0, stream>>>(xn2B, W1t, b1, nullptr, nullptr, hB, M, DFFq, Dq, 1);

    // 9) FFN2 + residual -> final f32 output
    dim3 gF2(Dq / 128, M / 128);
    gemm_bf16<<<gF2, 256, 0, stream>>>(hB, W2t, b2, x1F, (float*)d_out, nullptr, M, Dq, DFFq, 0);

    (void)in_sizes; (void)n_in; (void)out_size; (void)ws_size;
}
